// Decoder_53489522704422
// MI455X (gfx1250) — matrix operand order repaired and checked by reading, not compiled, not run on hardware
//
#include <hip/hip_runtime.h>
#include <cstdint>
#include <cstddef>

// ---------------------------------------------------------------------------
// MI455X decoder: batched WMMA GEMMs (double-buffered async-to-LDS staging)
// for the non-recurrent 78% of FLOPs, persistent 12-block grid-barrier kernel
// for the h@W_hh recurrence with W_hh resident in LDS, f16 WMMA w/ f32 accum.
// ---------------------------------------------------------------------------

typedef __attribute__((ext_vector_type(16))) _Float16 v16h;
typedef __attribute__((ext_vector_type(8)))  _Float16 v8h;
typedef __attribute__((ext_vector_type(8)))  float    v8f;
typedef int v4i_ __attribute__((vector_size(16)));

#define B_    128
#define T_    512
#define H_    384
#define DENC  768
#define HID_  768
#define POSD  128
#define LAB   128
#define XW    896     /* POSD + DENC */
#define G4H   1536

#if __has_builtin(__builtin_amdgcn_global_load_async_to_lds_b128)
#define HAVE_ASYNC 1
#else
#define HAVE_ASYNC 0
#endif

__device__ __forceinline__ float sigmoidf_(float x) { return 1.0f / (1.0f + __expf(-x)); }

// 16-byte global -> LDS copy (async on CDNA5, manual fallback otherwise)
__device__ __forceinline__ void cp16_g2l(const _Float16* g, _Float16* l)
{
#if HAVE_ASYNC
    __builtin_amdgcn_global_load_async_to_lds_b128(
        (__attribute__((address_space(1))) v4i_*)(__attribute__((address_space(1))) const void*)g,
        (__attribute__((address_space(3))) v4i_*)(__attribute__((address_space(3))) void*)l,
        0, 0);
#else
    v8h v = *(const v8h*)g;
    *(v8h*)l = v;
#endif
}

__device__ __forceinline__ void wait_async0()
{
#if HAVE_ASYNC
    asm volatile("s_wait_asynccnt 0" ::: "memory");
#endif
}

// ---------------------------------------------------------------------------
// K0a: f32 -> f16 weight conversion
// ---------------------------------------------------------------------------
__global__ __launch_bounds__(256)
void k_cvt(const float* __restrict__ src, _Float16* __restrict__ dst, int n)
{
    const int i = blockIdx.x * 256 + threadIdx.x;
    if (i < n) dst[i] = (_Float16)src[i];
}

// ---------------------------------------------------------------------------
// K0b: zero H row 0 and barrier state
// ---------------------------------------------------------------------------
__global__ __launch_bounds__(256)
void k_init(_Float16* __restrict__ H0, unsigned* __restrict__ bar)
{
    const int i = blockIdx.x * 256 + threadIdx.x;
    if (i < B_ * H_) H0[i] = (_Float16)0.0f;
    if (i < 2) bar[i] = 0u;
}

// ---------------------------------------------------------------------------
// K1: build X[m,896] = [pos_embed(pid) | avg(enc[s:t])], f16, m = t*B + b
// ---------------------------------------------------------------------------
__global__ __launch_bounds__(128)
void k_build_x(const float* __restrict__ enc, const float* __restrict__ posw,
               const int* __restrict__ wstart, const int* __restrict__ pids,
               _Float16* __restrict__ X)
{
    const int m = blockIdx.x;
    const int t = m >> 7, b = m & 127;
    const int k = threadIdx.x;              // 0..127
    const size_t xrow = (size_t)m * XW;

    const int pid = pids[t * B_ + b];
    X[xrow + k] = (_Float16)posw[(size_t)pid * POSD + k];

    const int st = wstart[t * B_ + b];
    const bool valid = (st >= 0);
    int s = st < 0 ? 0 : st;
    int len = t - s; if (len < 1) len = 1;
    float acc[6];
#pragma unroll
    for (int c = 0; c < 6; ++c) acc[c] = 0.0f;
    const float* eb = enc + (size_t)b * T_ * DENC;
    const int e = valid ? t : 0;
    for (int r = s; r < e; ++r) {
        const float* p = eb + (size_t)r * DENC + k;
#pragma unroll
        for (int c = 0; c < 6; ++c) acc[c] += p[c * 128];
    }
    const float inv = valid ? (1.0f / (float)len) : 0.0f;
#pragma unroll
    for (int c = 0; c < 6; ++c)
        X[xrow + POSD + k + c * 128] = (_Float16)(acc[c] * inv);
}

// ---------------------------------------------------------------------------
// WMMA GEMM template: C[M,N] = A[M,K] * B16[N,K]^T, 128x128 block tile,
// 8 waves (4M x 2N), each wave: 2 M-tiles x 4 N-tiles of 16x16.
// Double-buffered LDS tiles staged with GLOBAL_LOAD_ASYNC_TO_LDS_B128.
// MODE 0: combine  (A=X f16, K=896,  N=768,  out = f16 tanh(.+bias), t==0 rows zeroed)
// MODE 1: W_ih     (A=Z f16, K=768,  N=1536, out = f16)
// MODE 2: logits   (A=[Hbuf f16 | enc f32], K=1152, N=128, out = f32)
// ---------------------------------------------------------------------------
template <int MODE>
__global__ __launch_bounds__(256)
void k_gemm(const _Float16* __restrict__ A16, const float* __restrict__ encF,
            const _Float16* __restrict__ B16, const float* __restrict__ bias,
            _Float16* __restrict__ O16, float* __restrict__ O32)
{
    constexpr int K   = (MODE == 0) ? XW : (MODE == 1) ? HID_ : (H_ + DENC);
    constexpr int AW  = (MODE == 0) ? XW : (MODE == 1) ? HID_ : 0;
    constexpr int OST = (MODE == 0) ? HID_ : (MODE == 1) ? G4H : LAB;
    constexpr int NC  = K / 32;

    __shared__ _Float16 As[2][128][40];
    __shared__ _Float16 Bs[2][128][40];

    const int tid  = threadIdx.x;
    const int lane = tid & 31;
    const int wave = tid >> 5;
    const int wm = wave >> 1;      // 0..3
    const int wn = wave & 1;       // 0..1
    const int lm = lane & 15;
    const int hi8 = ((lane >> 4) & 1) * 8;
    const int Mb = blockIdx.x * 128;
    const int Nb = blockIdx.y * 128;

    // stage one 128x32 A tile + 128x32 B tile into LDS buffer `buf`
    auto stage = [&](int buf, int k0) {
        for (int q = tid; q < 512; q += 256) {
            const int row = q >> 2;
            const int c = (q & 3) << 3;
            if constexpr (MODE == 2) {
                const int m = Mb + row;
                if (k0 < H_) {  // h2 part: Hbuf row (t+1)*B + b == m + 128
                    cp16_g2l(A16 + ((size_t)(m + 128)) * H_ + k0 + c, &As[buf][row][c]);
                } else {        // encoder part (f32 -> f16, manual)
                    const int t = m >> 7, b = m & 127;
                    const float* p = encF + ((size_t)b * T_ + t) * DENC + (k0 - H_) + c;
#pragma unroll
                    for (int i = 0; i < 8; ++i) As[buf][row][c + i] = (_Float16)p[i];
                }
            } else {
                cp16_g2l(A16 + (size_t)(Mb + row) * AW + k0 + c, &As[buf][row][c]);
            }
        }
        for (int q = tid; q < 512; q += 256) {
            const int row = q >> 2;
            const int c = (q & 3) << 3;
            cp16_g2l(B16 + (size_t)(Nb + row) * K + k0 + c, &Bs[buf][row][c]);
        }
    };

    v8f acc[2][4];
#pragma unroll
    for (int mt = 0; mt < 2; ++mt)
#pragma unroll
        for (int nt = 0; nt < 4; ++nt)
#pragma unroll
            for (int i = 0; i < 8; ++i) acc[mt][nt][i] = 0.0f;

    stage(0, 0);

    for (int kc = 0; kc < NC; ++kc) {
        wait_async0();          // staged data for buffer kc&1 landed in LDS
        __syncthreads();        // visible to all waves; prior readers done
        if (kc + 1 < NC) stage((kc + 1) & 1, (kc + 1) * 32);

        const int buf = kc & 1;
        v16h af[2];
#pragma unroll
        for (int mt = 0; mt < 2; ++mt) {
            const int rowm = wm * 32 + mt * 16 + lm;
            const int off = (lane < 16) ? 0 : 8;
#pragma unroll
            for (int i = 0; i < 8; ++i) {
                af[mt][i]     = As[buf][rowm][off + i];
                af[mt][8 + i] = As[buf][rowm][off + 16 + i];
            }
        }
        v16h bf[4];
#pragma unroll
        for (int nt = 0; nt < 4; ++nt) {
            const int rown = wn * 64 + nt * 16 + lm;
            const int koff = (lane < 16) ? 0 : 8;
#pragma unroll
            for (int i = 0; i < 16; ++i) bf[nt][i] = Bs[buf][rown][koff + i + (i & 8)];
        }
#pragma unroll
        for (int mt = 0; mt < 2; ++mt)
#pragma unroll
            for (int nt = 0; nt < 4; ++nt)
                acc[mt][nt] = __builtin_amdgcn_wmma_f32_16x16x32_f16(
                    false, af[mt], false, bf[nt], (short)0, acc[mt][nt], false, false);
    }

    // ---- epilogue ----
#pragma unroll
    for (int mt = 0; mt < 2; ++mt)
#pragma unroll
        for (int nt = 0; nt < 4; ++nt) {
            const int gm0 = Mb + wm * 32 + mt * 16 + hi8;
            const int gn  = Nb + wn * 64 + nt * 16 + lm;
#pragma unroll
            for (int r = 0; r < 8; ++r) {
                const int gm = gm0 + r;
                float v = acc[mt][nt][r];
                if constexpr (MODE == 0) {
                    v = tanhf(v + bias[gn]);
                    if (gm < B_) v = 0.0f;          // t == 0 -> z = 0
                    O16[(size_t)gm * OST + gn] = (_Float16)v;
                } else if constexpr (MODE == 1) {
                    O16[(size_t)gm * OST + gn] = (_Float16)v;
                } else {
                    O32[(size_t)gm * OST + gn] = v;
                }
            }
        }
}

// ---------------------------------------------------------------------------
// K4: persistent recurrent kernel. 12 blocks, block n owns h-columns
// [32n, 32n+32). Per step: g = Gin[t] + bias + h @ W_hh^T (WMMA), gates in
// registers, h2 -> Hbuf[t+1], grid barrier. W_hh slice stays in LDS.
// ---------------------------------------------------------------------------
__global__ __launch_bounds__(256)
void k_recurrent(const _Float16* __restrict__ Gin, const float* __restrict__ Whh,
                 const float* __restrict__ bih, const float* __restrict__ bhh,
                 _Float16* __restrict__ Hbuf, unsigned* __restrict__ bar)
{
    extern __shared__ _Float16 Bw[];            // [128][392] f16 = 98KB
    const int tid  = threadIdx.x;
    const int lane = tid & 31;
    const int wave = tid >> 5;                  // M-tile index 0..7
    const int lm   = lane & 15;
    const int hi8  = ((lane >> 4) & 1) * 8;
    const int j0   = blockIdx.x * 32;

    // Stage W_hh slice once: row rr = gate*32 + jj  <->  Whh[gate*384 + j0+jj, :]
    for (int q = tid; q < 128 * 48; q += 256) {
        const int rr = q / 48;
        const int c  = (q % 48) * 8;
        const int g = rr >> 5, jj = rr & 31;
        const float* p = Whh + (size_t)(g * H_ + j0 + jj) * H_ + c;
#pragma unroll
        for (int i = 0; i < 8; ++i) Bw[rr * 392 + c + i] = (_Float16)p[i];
    }

    float bs[4][2];
#pragma unroll
    for (int g = 0; g < 4; ++g)
#pragma unroll
        for (int ns = 0; ns < 2; ++ns) {
            const int j = j0 + ns * 16 + lm;
            bs[g][ns] = bih[g * H_ + j] + bhh[g * H_ + j];
        }

    float cst[2][8];
#pragma unroll
    for (int ns = 0; ns < 2; ++ns)
#pragma unroll
        for (int r = 0; r < 8; ++r) cst[ns][r] = 0.0f;

    __syncthreads();

    for (int t = 0; t < T_; ++t) {
        // C-init = Gin + biases
        v8f acc[4][2];
#pragma unroll
        for (int g = 0; g < 4; ++g)
#pragma unroll
            for (int ns = 0; ns < 2; ++ns)
#pragma unroll
                for (int r = 0; r < 8; ++r) {
                    const int b = wave * 16 + r + hi8;
                    acc[g][ns][r] = (float)Gin[((size_t)t * B_ + b) * G4H +
                                               g * H_ + j0 + ns * 16 + lm] + bs[g][ns];
                }

        // prefetch next step's Gin slice for this block (global_prefetch_b8)
        if (t + 1 < T_) {
            const _Float16* pf = Gin + ((size_t)(t + 1) * B_ + (tid >> 1)) * G4H +
                                 (tid & 1) * (2 * H_) + j0;
            __builtin_prefetch((const void*)pf, 0, 1);
        }

        const _Float16* hrow = Hbuf + (size_t)t * B_ * H_;
        for (int k0 = 0; k0 < H_; k0 += 32) {
            v16h a;
            {
                const int b = wave * 16 + lm;
                const int off = (lane < 16) ? 0 : 8;
                const _Float16* p = hrow + (size_t)b * H_ + k0 + off;
                v8h v0 = *(const v8h*)p;
                v8h v1 = *(const v8h*)(p + 16);
#pragma unroll
                for (int i = 0; i < 8; ++i) { a[i] = v0[i]; a[8 + i] = v1[i]; }
            }
#pragma unroll
            for (int g = 0; g < 4; ++g)
#pragma unroll
                for (int ns = 0; ns < 2; ++ns) {
                    v16h bfr;
                    const int rr = g * 32 + ns * 16 + lm;
                    const int koff = (lane < 16) ? 0 : 8;
#pragma unroll
                    for (int i = 0; i < 16; ++i) bfr[i] = Bw[rr * 392 + k0 + koff + i + (i & 8)];
                    acc[g][ns] = __builtin_amdgcn_wmma_f32_16x16x32_f16(
                        false, a, false, bfr, (short)0, acc[g][ns], false, false);
                }
        }

        // gates + write h2
#pragma unroll
        for (int ns = 0; ns < 2; ++ns)
#pragma unroll
            for (int r = 0; r < 8; ++r) {
                const int b = wave * 16 + r + hi8;
                const int j = j0 + ns * 16 + lm;
                const float iv = sigmoidf_(acc[0][ns][r]);
                const float fv = sigmoidf_(acc[1][ns][r]);
                const float gv = tanhf(acc[2][ns][r]);
                const float ov = sigmoidf_(acc[3][ns][r]);
                const float c2 = fv * cst[ns][r] + iv * gv;
                cst[ns][r] = c2;
                const float h2 = ov * tanhf(c2);
                Hbuf[((size_t)(t + 1) * B_ + b) * H_ + j] = (_Float16)h2;
            }

        // grid barrier (sense via generation counter)
        __threadfence();
        __syncthreads();
        if (tid == 0) {
            const unsigned g = __atomic_load_n(bar + 1, __ATOMIC_RELAXED);
            const unsigned a = atomicAdd(bar, 1u);
            if (a == 11u) {
                __atomic_store_n(bar, 0u, __ATOMIC_RELAXED);
                __atomic_fetch_add(bar + 1, 1u, __ATOMIC_RELEASE);
            } else {
                while (__atomic_load_n(bar + 1, __ATOMIC_ACQUIRE) == g)
                    __builtin_amdgcn_s_sleep(1);
            }
        }
        __syncthreads();
    }
}

// ---------------------------------------------------------------------------
// K6: log_softmax over 128 labels; t==0 APP_ID fix; (t,b) -> row b*T + t
// ---------------------------------------------------------------------------
__global__ __launch_bounds__(128)
void k_logsoftmax(const float* __restrict__ L, float* __restrict__ out)
{
    __shared__ float red[128];
    const int m = blockIdx.x;
    const int t = m >> 7, b = m & 127;
    const int l = threadIdx.x;
    float x = L[(size_t)m * LAB + l];
    if (t == 0 && l == 3) x = -1e10f;
    red[l] = x; __syncthreads();
#pragma unroll
    for (int s = 64; s > 0; s >>= 1) {
        if (l < s) red[l] = fmaxf(red[l], red[l + s]);
        __syncthreads();
    }
    const float mx = red[0]; __syncthreads();
    const float e = __expf(x - mx);
    red[l] = e; __syncthreads();
#pragma unroll
    for (int s = 64; s > 0; s >>= 1) {
        if (l < s) red[l] += red[l + s];
        __syncthreads();
    }
    const float lse = __logf(red[0]);
    out[((size_t)b * T_ + t) * LAB + l] = x - mx - lse;
}

// ---------------------------------------------------------------------------
extern "C" void kernel_launch(void* const* d_in, const int* in_sizes, int n_in,
                              void* d_out, int out_size, void* d_ws, size_t ws_size,
                              hipStream_t stream)
{
    (void)in_sizes; (void)n_in; (void)out_size; (void)ws_size;
    const float* enc    = (const float*)d_in[0];
    const float* posw   = (const float*)d_in[1];
    const float* W_ih   = (const float*)d_in[2];
    const float* W_hh   = (const float*)d_in[3];
    const float* b_ih   = (const float*)d_in[4];
    const float* b_hh   = (const float*)d_in[5];
    const float* comb_W = (const float*)d_in[6];
    const float* comb_b = (const float*)d_in[7];
    const float* out_W  = (const float*)d_in[8];
    const int*   wstart = (const int*)d_in[9];
    const int*   pids   = (const int*)d_in[10];
    float* out = (float*)d_out;

    char* ws = (char*)d_ws;
    const size_t M = (size_t)T_ * B_;                 // 65536
    // workspace layout (Lbuf aliases X: X is dead after the combine GEMM)
    const size_t offX   = 0;
    const size_t offZ   = offX + M * XW * 2;                    // 117,440,512
    const size_t offG   = offZ + M * HID_ * 2;                  // +100,663,296
    const size_t offH   = offG + M * G4H * 2;                   // +201,326,592
    const size_t offBar = offH + (size_t)(T_ + 1) * B_ * H_ * 2;
    const size_t offWc  = (offBar + 255 + 8) & ~(size_t)255;
    const size_t offWi  = offWc + (size_t)HID_ * XW * 2;
    const size_t offWo  = offWi + (size_t)G4H * HID_ * 2;

    _Float16* X    = (_Float16*)(ws + offX);
    float*    Lbuf = (float*)(ws + offX);             // aliases X
    _Float16* Z    = (_Float16*)(ws + offZ);
    _Float16* Gin  = (_Float16*)(ws + offG);
    _Float16* Hbuf = (_Float16*)(ws + offH);
    unsigned* bar  = (unsigned*)(ws + offBar);
    _Float16* Wc16 = (_Float16*)(ws + offWc);
    _Float16* Wi16 = (_Float16*)(ws + offWi);
    _Float16* Wo16 = (_Float16*)(ws + offWo);

    // K0: zero h0 + barrier; convert weights to f16
    k_init<<<192, 256, 0, stream>>>(Hbuf, bar);
    k_cvt<<<(HID_ * XW + 255) / 256, 256, 0, stream>>>(comb_W, Wc16, HID_ * XW);
    k_cvt<<<(G4H * HID_ + 255) / 256, 256, 0, stream>>>(W_ih, Wi16, G4H * HID_);
    k_cvt<<<(LAB * (H_ + DENC) + 255) / 256, 256, 0, stream>>>(out_W, Wo16, LAB * (H_ + DENC));
    // K1: build X
    k_build_x<<<(int)M, 128, 0, stream>>>(enc, posw, wstart, pids, X);
    // K2: Z = tanh(X @ combine_W^T + comb_b), t==0 rows zeroed
    k_gemm<0><<<dim3(512, 6), 256, 0, stream>>>(X, nullptr, Wc16, comb_b, Z, nullptr);
    // K3: Gin = Z @ W_ih^T
    k_gemm<1><<<dim3(512, 12), 256, 0, stream>>>(Z, nullptr, Wi16, nullptr, Gin, nullptr);
    // K4: sequential LSTM recurrence (persistent, grid barrier per step)
    k_recurrent<<<12, 256, 128 * 392 * sizeof(_Float16), stream>>>(
        Gin, W_hh, b_ih, b_hh, Hbuf, bar);
    // K5: Lbuf = [h2 | e_t] @ out_W^T
    k_gemm<2><<<dim3(512, 1), 256, 0, stream>>>(Hbuf, enc, Wo16, nullptr, nullptr, Lbuf);
    // K6: log_softmax + t==0 APP_ID clamp + row remap to (b*T + t)
    k_logsoftmax<<<(int)M, 128, 0, stream>>>(Lbuf, out);
}
